// GraphSelfAttentionLayer_17420387352923
// MI455X (gfx1250) — matrix operand order repaired and checked by reading, not compiled, not run on hardware
//
#include <hip/hip_runtime.h>
#include <hip/hip_bf16.h>

// ---------------------------------------------------------------------------
// Problem constants
// ---------------------------------------------------------------------------
constexpr int Bc  = 16;            // batch
constexpr int Nn  = 300;           // queries
constexpr int NP  = 304;           // N padded to 16
constexpr int NG  = 150;           // keys / groups
constexpr int NGP = 160;           // NG padded to 32 (K-dim of WMMA)
constexpr int Dd  = 1024;          // model dim
constexpr int Hh  = 16;            // heads
constexpr int Pp  = 64;            // pos-emb dim
constexpr int HD  = 64;            // head dim
constexpr int Rr  = Nn * NG;       // 45000 pos rows per batch
constexpr int RT  = (Rr + 15) / 16;// 2813 row tiles

// ---------------------------------------------------------------------------
// Workspace layout (bytes). Total ~107 MB.
// ---------------------------------------------------------------------------
constexpr size_t o_roi16  = 0;                                   // bf16 roi, n-padded  [B][NP][D]
constexpr size_t s_roi16  = (size_t)Bc * NP * Dd * 2;
constexpr size_t o_wqT    = o_roi16 + s_roi16;                   // bf16 Wq^T [D][D]
constexpr size_t s_wT     = (size_t)Dd * Dd * 2;
constexpr size_t o_wkT    = o_wqT + s_wT;                        // bf16 Wk^T [D][D]
constexpr size_t o_wpT    = o_wkT + s_wT;                        // bf16 Wp^T [H][P]
constexpr size_t s_wpT    = 4096;
constexpr size_t o_woutT  = o_wpT + s_wpT;                       // bf16 Wout^T [H][HD][D]
constexpr size_t s_woutT  = (size_t)Hh * HD * Dd * 2;
constexpr size_t o_q16    = o_woutT + s_woutT;                   // bf16 q [B][NP][D]
constexpr size_t s_q16    = (size_t)Bc * NP * Dd * 2;
constexpr size_t o_k16    = o_q16 + s_q16;                       // bf16 k [B][NGP][D]
constexpr size_t s_k16    = (size_t)Bc * NGP * Dd * 2;
constexpr size_t o_truncT = o_k16 + s_k16;                       // bf16 trunc^T [B][D][NGP]
constexpr size_t s_truncT = (size_t)Bc * Dd * NGP * 2;
constexpr size_t o_waff   = o_truncT + s_truncT;                 // f32 waff [B][H][NP][NGP]
constexpr size_t s_waff   = (size_t)Bc * Hh * NP * NGP * 4;
constexpr size_t o_attn16 = o_waff + s_waff;                     // bf16 attn [B][H][NP][NGP]

// ---------------------------------------------------------------------------
// WMMA plumbing (gfx1250, wave32, V_WMMA_F32_16X16X32_BF16)
// ---------------------------------------------------------------------------
typedef __attribute__((ext_vector_type(16))) __bf16        bf16x16;
typedef __attribute__((ext_vector_type(8)))  float         f32x8;
typedef __attribute__((ext_vector_type(4)))  unsigned int  u32x4;
typedef __attribute__((ext_vector_type(4)))  int           i32x4;
typedef __attribute__((ext_vector_type(8)))  int           i32x8;
typedef __attribute__((ext_vector_type(4)))  float         f32x4;

union FragB {
    bf16x16        v;
    u32x4          q[2];
    unsigned short us[16];
};

__device__ __forceinline__ unsigned short f2bf(float f) {
    unsigned int u = __float_as_uint(f);
    unsigned int r = (u + 0x7FFFu + ((u >> 16) & 1u)) >> 16;   // round-nearest-even
    return (unsigned short)r;
}

__device__ __forceinline__ f32x8 zero8() {
    f32x8 z = {0.f, 0.f, 0.f, 0.f, 0.f, 0.f, 0.f, 0.f};
    return z;
}

__device__ __forceinline__ f32x8 wmma_bf16(const FragB& a, const FragB& b, f32x8 c) {
    return __builtin_amdgcn_wmma_f32_16x16x32_bf16(
        /*neg_a=*/false, a.v, /*neg_b=*/false, b.v,
        /*c_mod=*/(short)0, c, /*reuse_a=*/false, /*reuse_b=*/false);
}

// A fragment (16x32, row-major source, 32-contiguous K): lane m = L%16,
// VGPR0-3 <- K = 8*half+0..7, VGPR4-7 <- K = 16+8*half+0..7  (ISA 7.12.2)
__device__ __forceinline__ FragB load_a_frag(const unsigned short* base, int row0,
                                             int ld, int k0, int lane) {
    FragB f;
    int m = lane & 15, hl = lane >> 4;
    const unsigned short* p = base + (size_t)(row0 + m) * ld + k0 + 8 * hl;
    f.q[0] = *reinterpret_cast<const u32x4*>(p);
    f.q[1] = *reinterpret_cast<const u32x4*>(p + 16);
    return f;
}

// B fragment (32x16) loaded from B^T (row-major, rows = output column):
// lane n = L%16, VGPR0-3 <- K = 8*half+0..7, VGPR4-7 <- K = 16+8*half+0..7
__device__ __forceinline__ FragB load_bt_frag(const unsigned short* base, int col0,
                                              int ld, int k0, int lane) {
    FragB f;
    int n = lane & 15, hl = lane >> 4;
    const unsigned short* p = base + (size_t)(col0 + n) * ld + k0 + 8 * hl;
    f.q[0] = *reinterpret_cast<const u32x4*>(p);
    f.q[1] = *reinterpret_cast<const u32x4*>(p + 16);
    return f;
}

// ---------------------------------------------------------------------------
// Tensor Data Mover: 2D tile (rows x NGP bf16 elements, row stride NGP) from
// global -> LDS. D# packed per CDNA5 ISA 8.3/8.4 (group0: count/lds/global/
// type=2; group1: data_size=2B, tensor+tile dims, dim0 stride).
// ---------------------------------------------------------------------------
#if __has_builtin(__builtin_amdgcn_tensor_load_to_lds)
#define HAVE_TDM 1
__device__ __forceinline__ void tdm_load_tile_2d(unsigned int lds_addr,
                                                 unsigned long long gaddr,
                                                 int rows) {
    u32x4 g0;
    g0[0] = 1u;                                             // count=1, user D#
    g0[1] = lds_addr;                                       // lds_addr [63:32]
    g0[2] = (unsigned int)gaddr;                            // global_addr lo
    g0[3] = ((unsigned int)(gaddr >> 32) & 0x01FFFFFFu)     // global_addr hi
          | (2u << 30);                                     // type=2 (image)
    i32x8 g1;
    g1[0] = (int)(1u << 16);                 // workgroup_mask=0, data_size=1 (2B)
    g1[1] = (int)((unsigned)NGP << 16);      // tensor_dim0 lo16 -> bits[63:48]
    g1[2] = (int)((unsigned)rows << 16);     // tensor_dim0 hi=0 | tensor_dim1 lo16
    g1[3] = (int)((unsigned)NGP << 16);      // tensor_dim1 hi=0 | tile_dim0=NGP
    g1[4] = rows;                            // tile_dim1=rows | tile_dim2=0
    g1[5] = NGP;                             // tensor_dim0_stride lo32
    g1[6] = 0;                               // stride0 hi | stride1 lo
    g1[7] = 0;                               // stride1 hi
    i32x4 gz = {0, 0, 0, 0};
#if defined(__clang_major__) && (__clang_major__ >= 23)
    i32x8 gz8 = {0, 0, 0, 0, 0, 0, 0, 0};
    __builtin_amdgcn_tensor_load_to_lds(g0, g1, gz, gz, gz8, 0);
#else
    __builtin_amdgcn_tensor_load_to_lds(g0, g1, gz, gz, 0);
#endif
}
#else
#define HAVE_TDM 0
#endif

// ---------------------------------------------------------------------------
// Prep kernels: bf16 conversions / transposes / padding
// ---------------------------------------------------------------------------
__global__ void k_transpose_d(const float* __restrict__ W, unsigned short* __restrict__ WT) {
    int i = blockIdx.x * 256 + threadIdx.x;          // over D*D
    int n = i >> 10, k = i & 1023;
    WT[(size_t)n * Dd + k] = f2bf(W[(size_t)k * Dd + n]);
}

__global__ void k_prep_small(const float* __restrict__ Wp, const float* __restrict__ Wout,
                             unsigned short* __restrict__ WpT, unsigned short* __restrict__ WoutT) {
    int i = blockIdx.x * 256 + threadIdx.x;          // over H*HD*D
    if (i < Hh * Pp) {                               // WpT[h][p] = Wp[p][h]
        int h = i >> 6, p = i & 63;
        WpT[i] = f2bf(Wp[p * Hh + h]);
    }
    int h = i >> 16;                                 // WoutT[h][o][d] = Wout[h][d][o]
    int o = (i >> 10) & 63;
    int d = i & 1023;
    WoutT[i] = f2bf(Wout[((size_t)h * Dd + d) * HD + o]);
}

__global__ void k_prep_roi(const float* __restrict__ roi, unsigned short* __restrict__ roi16) {
    int i = blockIdx.x * 256 + threadIdx.x;          // over B*NP*D
    int b = i / (NP * Dd);
    int rem = i - b * (NP * Dd);
    int n = rem >> 10, d = rem & 1023;
    float v = (n < Nn) ? roi[((size_t)b * Nn + n) * Dd + d] : 0.f;
    roi16[i] = f2bf(v);
}

__global__ void k_prep_truncT(const float* __restrict__ roi, unsigned short* __restrict__ tT) {
    int i = blockIdx.x * 256 + threadIdx.x;          // over B*D*NGP
    int b = i / (Dd * NGP);
    int rem = i - b * (Dd * NGP);
    int d = rem / NGP, m = rem - d * NGP;
    float v = (m < NG) ? roi[((size_t)b * Nn + m) * Dd + d] : 0.f;
    tT[i] = f2bf(v);
}

// ---------------------------------------------------------------------------
// Linear GEMM: C[b] = A[b] @ W + bias  (A bf16 rows, W^T bf16), out bf16.
// Block 256 = 8 waves; each wave computes a 16x64 tile, K-loop 1024/32,
// software-pipelined: k+1 fragments are in flight during k's WMMAs.
// ---------------------------------------------------------------------------
__global__ void __launch_bounds__(256)
k_gemm_lin(const unsigned short* __restrict__ A, long aStride,
           const unsigned short* __restrict__ BT, const float* __restrict__ bias,
           unsigned short* __restrict__ C, long cStride) {
    int lane = threadIdx.x & 31, wave = threadIdx.x >> 5, hl = lane >> 4;
    const unsigned short* Ab = A + (size_t)blockIdx.z * aStride;
    unsigned short*       Cb = C + (size_t)blockIdx.z * cStride;
    int row0  = blockIdx.x * 16;
    int ncol0 = (blockIdx.y * 8 + wave) * 64;

    f32x8 acc[4];
    #pragma unroll
    for (int t = 0; t < 4; ++t) acc[t] = zero8();

    // prologue: fragments for k0 = 0
    FragB a_cur = load_a_frag(Ab, row0, Dd, 0, lane);
    FragB b_cur[4];
    #pragma unroll
    for (int t = 0; t < 4; ++t) b_cur[t] = load_bt_frag(BT, ncol0 + t * 16, Dd, 0, lane);

    #pragma unroll 4
    for (int k0 = 0; k0 < Dd; k0 += 32) {
        FragB a_nxt = a_cur;
        FragB b_nxt[4];
        int kn = k0 + 32;
        if (kn < Dd) {
            __builtin_prefetch(Ab + (size_t)(row0 + (lane & 15)) * Dd + kn + 32, 0, 0);
            a_nxt = load_a_frag(Ab, row0, Dd, kn, lane);
            #pragma unroll
            for (int t = 0; t < 4; ++t)
                b_nxt[t] = load_bt_frag(BT, ncol0 + t * 16, Dd, kn, lane);
        } else {
            #pragma unroll
            for (int t = 0; t < 4; ++t) b_nxt[t] = b_cur[t];
        }
        #pragma unroll
        for (int t = 0; t < 4; ++t)
            acc[t] = wmma_bf16(a_cur, b_cur[t], acc[t]);
        a_cur = a_nxt;
        #pragma unroll
        for (int t = 0; t < 4; ++t) b_cur[t] = b_nxt[t];
    }

    #pragma unroll
    for (int t = 0; t < 4; ++t) {
        int col = ncol0 + t * 16 + (lane & 15);
        float bv = bias[col];
        #pragma unroll
        for (int r = 0; r < 8; ++r) {
            int row = row0 + r + 8 * hl;
            Cb[(size_t)row * Dd + col] = f2bf(acc[t][r] + bv);
        }
    }
}

// ---------------------------------------------------------------------------
// waff[b,h,n,m] = scale * q[b,n,h,:] . k[b,m,h,:]   (one wave: 16n x 160m)
// ---------------------------------------------------------------------------
__global__ void k_gemm_waff(const unsigned short* __restrict__ q16,
                            const unsigned short* __restrict__ k16,
                            float* __restrict__ waff) {
    int lane = threadIdx.x & 31, hl = lane >> 4;
    int bh = blockIdx.y, b = bh >> 4, h = bh & 15;
    int row0 = blockIdx.x * 16;
    const unsigned short* Ab = q16 + (size_t)b * NP * Dd + h * HD;
    const unsigned short* Bb = k16 + (size_t)b * NGP * Dd + h * HD;

    f32x8 acc[10];
    #pragma unroll
    for (int t = 0; t < 10; ++t) acc[t] = zero8();

    #pragma unroll
    for (int kt = 0; kt < 2; ++kt) {
        int k0 = kt * 32;
        FragB a = load_a_frag(Ab, row0, Dd, k0, lane);
        #pragma unroll
        for (int mt = 0; mt < 10; ++mt) {
            FragB bb = load_bt_frag(Bb, mt * 16, Dd, k0, lane);
            acc[mt] = wmma_bf16(a, bb, acc[mt]);
        }
    }
    const float scale = 0.125f;                      // 1/sqrt(64)
    float* Wb = waff + (size_t)bh * NP * NGP;
    #pragma unroll
    for (int mt = 0; mt < 10; ++mt)
        #pragma unroll
        for (int r = 0; r < 8; ++r)
            Wb[(size_t)(row0 + r + 8 * hl) * NGP + mt * 16 + (lane & 15)] =
                acc[mt][r] * scale;
}

// ---------------------------------------------------------------------------
// pos GEMM: relu(pos_emb @ Wp + bp) -> log(max(.,1e-6)) scatter-added to waff.
// A converted f32->bf16 on the fly (one wave per 16 rows).
// ---------------------------------------------------------------------------
__global__ void k_gemm_pos(const float* __restrict__ pos_emb,
                           const unsigned short* __restrict__ WpT,
                           const float* __restrict__ bp,
                           float* __restrict__ waff) {
    int lane = threadIdx.x & 31, hl = lane >> 4;
    int rtile = blockIdx.x, b = blockIdx.y;
    int r = rtile * 16 + (lane & 15);
    int rs = (r < Rr) ? r : (Rr - 1);
    const float* pa = pos_emb + ((size_t)b * Rr + rs) * Pp;

    f32x8 acc = zero8();
    #pragma unroll
    for (int kt = 0; kt < 2; ++kt) {
        int k0 = kt * 32;
        FragB a;
        #pragma unroll
        for (int c2 = 0; c2 < 2; ++c2) {
            const float* p = pa + k0 + 16 * c2 + 8 * hl;
            f32x4 x0 = *reinterpret_cast<const f32x4*>(p);
            f32x4 x1 = *reinterpret_cast<const f32x4*>(p + 4);
            a.us[8 * c2 + 0] = f2bf(x0.x); a.us[8 * c2 + 1] = f2bf(x0.y);
            a.us[8 * c2 + 2] = f2bf(x0.z); a.us[8 * c2 + 3] = f2bf(x0.w);
            a.us[8 * c2 + 4] = f2bf(x1.x); a.us[8 * c2 + 5] = f2bf(x1.y);
            a.us[8 * c2 + 6] = f2bf(x1.z); a.us[8 * c2 + 7] = f2bf(x1.w);
        }
        FragB bb = load_bt_frag(WpT, 0, Pp, k0, lane);
        acc = wmma_bf16(a, bb, acc);
    }
    int h = lane & 15;
    float bv = bp[h];
    #pragma unroll
    for (int rr = 0; rr < 8; ++rr) {
        int row = rtile * 16 + rr + 8 * hl;
        if (row < Rr) {
            float v = acc[rr] + bv;
            v = v > 0.f ? v : 0.f;                   // relu
            v = __logf(fmaxf(v, 1e-6f));             // log(max(.,1e-6))
            int n = row / NG;
            int m = row - n * NG;
            waff[((size_t)(b * Hh + h) * NP + n) * NGP + m] += v;
        }
    }
}

// ---------------------------------------------------------------------------
// Masked softmax over m (one wave per (b,h,n) row), bf16 attn out, m-pads zeroed.
// ---------------------------------------------------------------------------
__global__ void k_softmax(const float* __restrict__ waff, const int* __restrict__ adj,
                          const float* __restrict__ label, unsigned short* __restrict__ attn16) {
    int lane = threadIdx.x;                          // blockDim = 32
    int n = blockIdx.x, bh = blockIdx.y, b = bh >> 4;
    const float* wrow = waff + ((size_t)bh * NP + n) * NGP;
    const int*   arow = adj   + ((size_t)b * Nn + n) * NG;
    const float* lrow = label + ((size_t)b * Nn + n) * NG;

    float x[5];
    float vmax = -3.4e38f;
    #pragma unroll
    for (int i = 0; i < 5; ++i) {
        int m = lane + 32 * i;
        if (m < NG) {
            float la = lrow[m];
            float v  = (arow[m] > 0) ? (wrow[m] + la) : (-9.0e15f + la);
            x[i] = v;
            vmax = fmaxf(vmax, v);
        } else x[i] = -3.4e38f;
    }
    #pragma unroll
    for (int off = 16; off >= 1; off >>= 1)
        vmax = fmaxf(vmax, __shfl_xor(vmax, off, 32));

    float e[5];
    float sum = 0.f;
    #pragma unroll
    for (int i = 0; i < 5; ++i) {
        int m = lane + 32 * i;
        if (m < NG) { e[i] = __expf(x[i] - vmax); sum += e[i]; }
        else e[i] = 0.f;
    }
    #pragma unroll
    for (int off = 16; off >= 1; off >>= 1)
        sum += __shfl_xor(sum, off, 32);
    float inv = 1.0f / sum;

    unsigned short* orow = attn16 + ((size_t)bh * NP + n) * NGP;
    #pragma unroll
    for (int i = 0; i < 5; ++i) {
        int m = lane + 32 * i;
        if (m < NG) orow[m] = f2bf(e[i] * inv);
    }
    if (lane < NGP - NG) orow[NG + lane] = 0;        // zero K-pads
}

// ---------------------------------------------------------------------------
// Fused: S = attn @ trunc (16x1024 per block, wave owns 128 cols), then
// O = S @ Wout[h]^T (per-wave K-slice partials reduced in LDS). The 16x160
// attn tile is staged to LDS once per block via the Tensor Data Mover
// (tensor_load_to_lds + s_wait_tensorcnt) and shared by all 8 waves.
// LDS: S slab 32KB + (Atile overlaid on Opart) 32KB = 64KB.
// ---------------------------------------------------------------------------
__global__ void __launch_bounds__(256)
k_attn_out(const unsigned short* __restrict__ attn16, const unsigned short* __restrict__ truncT,
           const unsigned short* __restrict__ woutT, const float* __restrict__ bout,
           float* __restrict__ out) {
    __shared__ __align__(16) unsigned char smemS[16 * 1024 * 2];   // bf16 S slab
    __shared__ __align__(16) unsigned char smemO[8 * 16 * 64 * 4]; // Opart / Atile

    unsigned short (*Slds)[1024] = reinterpret_cast<unsigned short(*)[1024]>(smemS);
    float (*Opart)[16][64]       = reinterpret_cast<float(*)[16][64]>(smemO);
    unsigned short* Atile        = reinterpret_cast<unsigned short*>(smemO);

    int lane = threadIdx.x & 31, wave = threadIdx.x >> 5, hl = lane >> 4;
    int bh = blockIdx.y, b = bh >> 4, h = bh & 15;
    int ntile = blockIdx.x;

    const unsigned short* Ab = attn16 + ((size_t)bh * NP + ntile * 16) * NGP;
    const unsigned short* Tb = truncT + (size_t)b * Dd * NGP;

    // ---- stage 0: TDM-copy the 16x160 attn tile into LDS ------------------
#if HAVE_TDM
    if (wave == 0) {
        tdm_load_tile_2d((unsigned int)(uintptr_t)Atile,
                         (unsigned long long)(uintptr_t)Ab, 16);
        __builtin_amdgcn_s_wait_tensorcnt(0);
    }
#else
    for (int i = threadIdx.x; i < (16 * NGP) / 2; i += 256)
        reinterpret_cast<unsigned int*>(Atile)[i] =
            reinterpret_cast<const unsigned int*>(Ab)[i];
#endif
    __syncthreads();

    // ---- stage 1: S[16][wave*128 .. +128) = attn_tile @ trunc -------------
    f32x8 s[8];
    #pragma unroll
    for (int t = 0; t < 8; ++t) s[t] = zero8();

    #pragma unroll
    for (int kt = 0; kt < 5; ++kt) {                 // K = m, 160/32
        int k0 = kt * 32;
        FragB a = load_a_frag(Atile, 0, NGP, k0, lane);   // ds_load path
        #pragma unroll
        for (int t = 0; t < 8; ++t) {
            int dcol = wave * 128 + t * 16;
            FragB bb = load_bt_frag(Tb, dcol, NGP, k0, lane);
            s[t] = wmma_bf16(a, bb, s[t]);
        }
    }
    #pragma unroll
    for (int t = 0; t < 8; ++t) {
        int dcol = wave * 128 + t * 16 + (lane & 15);
        #pragma unroll
        for (int r = 0; r < 8; ++r)
            Slds[r + 8 * hl][dcol] = f2bf(s[t][r]);
    }
    __syncthreads();   // Atile reads complete; smemO may now be reused as Opart

    // ---- stage 2: O_partial = S[:, wave slice] @ WoutT[h] -----------------
    const unsigned short* Wb = woutT + (size_t)h * HD * Dd;
    f32x8 o4[4];
    #pragma unroll
    for (int t = 0; t < 4; ++t) o4[t] = zero8();

    #pragma unroll
    for (int kt = 0; kt < 4; ++kt) {                 // K = d slice, 128/32
        int k0 = wave * 128 + kt * 32;
        FragB a = load_a_frag(&Slds[0][0], 0, 1024, k0, lane);  // ds_load path
        #pragma unroll
        for (int t = 0; t < 4; ++t) {
            FragB bb = load_bt_frag(Wb, t * 16, Dd, k0, lane);
            o4[t] = wmma_bf16(a, bb, o4[t]);
        }
    }
    #pragma unroll
    for (int t = 0; t < 4; ++t)
        #pragma unroll
        for (int r = 0; r < 8; ++r)
            Opart[wave][r + 8 * hl][t * 16 + (lane & 15)] = o4[t][r];
    __syncthreads();

    // ---- reduce 8 wave partials, add bias, store --------------------------
    int tid = threadIdx.x;
    #pragma unroll
    for (int j = 0; j < 4; ++j) {
        int e = tid * 4 + j;
        int rr = e >> 6, o = e & 63;
        float sum = 0.f;
        #pragma unroll
        for (int w = 0; w < 8; ++w) sum += Opart[w][rr][o];
        sum += bout[h * HD + o];
        int n = ntile * 16 + rr;
        if (n < Nn)
            out[((size_t)(b * Nn + n)) * Dd + h * HD + o] = sum;
    }
}

// ---------------------------------------------------------------------------
// Launch
// ---------------------------------------------------------------------------
extern "C" void kernel_launch(void* const* d_in, const int* in_sizes, int n_in,
                              void* d_out, int out_size, void* d_ws, size_t ws_size,
                              hipStream_t stream) {
    (void)in_sizes; (void)n_in; (void)out_size; (void)ws_size;
    const float* roi   = (const float*)d_in[0];
    const int*   adj   = (const int*)  d_in[1];
    const float* pos   = (const float*)d_in[2];
    const float* label = (const float*)d_in[3];
    const float* Wq    = (const float*)d_in[4];
    const float* bq    = (const float*)d_in[5];
    const float* Wk    = (const float*)d_in[6];
    const float* bk    = (const float*)d_in[7];
    const float* Wp    = (const float*)d_in[8];
    const float* bp    = (const float*)d_in[9];
    const float* Wout  = (const float*)d_in[10];
    const float* bout  = (const float*)d_in[11];
    float* out = (float*)d_out;

    char* ws = (char*)d_ws;
    unsigned short* roi16  = (unsigned short*)(ws + o_roi16);
    unsigned short* wqT    = (unsigned short*)(ws + o_wqT);
    unsigned short* wkT    = (unsigned short*)(ws + o_wkT);
    unsigned short* wpT    = (unsigned short*)(ws + o_wpT);
    unsigned short* woutT  = (unsigned short*)(ws + o_woutT);
    unsigned short* q16    = (unsigned short*)(ws + o_q16);
    unsigned short* k16    = (unsigned short*)(ws + o_k16);
    unsigned short* truncT = (unsigned short*)(ws + o_truncT);
    float*          waff   = (float*)(ws + o_waff);
    unsigned short* attn16 = (unsigned short*)(ws + o_attn16);

    // prep: bf16 conversions / transposes
    k_transpose_d <<<(Dd * Dd) / 256, 256, 0, stream>>>(Wq, wqT);
    k_transpose_d <<<(Dd * Dd) / 256, 256, 0, stream>>>(Wk, wkT);
    k_prep_small  <<<(Hh * HD * Dd) / 256, 256, 0, stream>>>(Wp, Wout, wpT, woutT);
    k_prep_roi    <<<(Bc * NP * Dd) / 256, 256, 0, stream>>>(roi, roi16);
    k_prep_truncT <<<(Bc * Dd * NGP) / 256, 256, 0, stream>>>(roi, truncT);

    // q = roi@Wq+bq (M=304 tiles=19), k = trunc@Wk+bk (M=160 tiles=10)
    k_gemm_lin<<<dim3(NP / 16, 2, Bc), 256, 0, stream>>>(
        roi16, (long)NP * Dd, wqT, bq, q16, (long)NP * Dd);
    k_gemm_lin<<<dim3(NGP / 16, 2, Bc), 256, 0, stream>>>(
        roi16, (long)NP * Dd, wkT, bk, k16, (long)NGP * Dd);

    // waff = scale * q k^T per head
    k_gemm_waff<<<dim3(NP / 16, Bc * Hh), 32, 0, stream>>>(q16, k16, waff);

    // waff += log(max(relu(pos_emb@Wp+bp),1e-6))
    k_gemm_pos<<<dim3(RT, Bc), 32, 0, stream>>>(pos, wpT, bp, waff);

    // masked softmax -> bf16 attn
    k_softmax<<<dim3(Nn, Bc * Hh), 32, 0, stream>>>(waff, adj, label, attn16);

    // fused (attn @ trunc) @ Wout[h] + bout -> out
    k_attn_out<<<dim3(NP / 16, Bc * Hh), 256, 0, stream>>>(attn16, truncT, woutT, bout, out);
}
